// FPSNet_89713276879421
// MI455X (gfx1250) — matrix operand order repaired and checked by reading, not compiled, not run on hardware
//
#include <hip/hip_runtime.h>
#include <hip/hip_bf16.h>

// ---------------------------------------------------------------------------
// Types for CDNA5 WMMA (wave32)
// ---------------------------------------------------------------------------
typedef __attribute__((ext_vector_type(16))) __bf16 v16bf;
typedef __attribute__((ext_vector_type(8)))  float  v8f;

union Frag16 { uint4 q[2]; unsigned short u[16]; v16bf v; };

__device__ __forceinline__ unsigned short f2bf(float f) {
  unsigned int u = __float_as_uint(f);
  unsigned int r = (u + 0x7FFFu + ((u >> 16) & 1u)) >> 16;   // RNE
  return (unsigned short)r;
}

// ---------------------------------------------------------------------------
// Weight packing into the exact CDNA5 A-fragment layout so conv waves read
// their A fragments straight from global memory (L2-resident, fully
// coalesced b128s), no LDS staging for A.
// Layout: dst[((s*8 + mtile)*32 + lane)*16 + j], j = per-lane half index.
//   m = mtile*16 + (lane&15); khalf = (lane>>4)*8
//   VGPR t = j>>1:  t<4 -> kb = khalf + 2t + (j&1)
//                   t>=4 -> kb = 16 + khalf + 2(t-4) + (j&1)
//   k = s*32 + kb
// ---------------------------------------------------------------------------
__global__ void pack_weights_kernel(const float* __restrict__ w,
                                    unsigned short* __restrict__ dst,
                                    int Kreal, int Cout, int total) {
  int idx = blockIdx.x * blockDim.x + threadIdx.x;
  if (idx >= total) return;
  int j    = idx & 15;
  int lane = (idx >> 4) & 31;
  int mt   = (idx >> 9) & 7;
  int s    = idx >> 12;
  int khalf = (lane >> 4) << 3;
  int m  = (mt << 4) + (lane & 15);
  int t  = j >> 1;
  int j1 = j & 1;
  int kb = (t < 4) ? (khalf + 2 * t + j1) : (16 + khalf + 2 * (t - 4) + j1);
  int k  = (s << 5) + kb;
  unsigned short v = 0;
  if (k < Kreal && m < Cout) v = f2bf(w[(size_t)m * Kreal + k]);
  dst[idx] = v;
}

// ---------------------------------------------------------------------------
// BN inference affine precompute: scale = g/sqrt(v+eps), shift = b - m*scale
// ---------------------------------------------------------------------------
__global__ void bn_prep_kernel(const float* __restrict__ gamma,
                               const float* __restrict__ beta,
                               const float* __restrict__ mean,
                               const float* __restrict__ var,
                               float* __restrict__ scale,
                               float* __restrict__ shift, int C) {
  int c = blockIdx.x * blockDim.x + threadIdx.x;
  if (c < C) {
    float s = gamma[c] * rsqrtf(var[c] + 1e-5f);
    scale[c] = s;
    shift[c] = beta[c] - mean[c] * s;
  }
}

// ---------------------------------------------------------------------------
// Implicit-GEMM conv (3x3 SAME or 1x1) via v_wmma_f32_16x16x32_bf16.
// 256 threads = 8 waves. Block tile: M=128 (Cout pad) x N=64 pixels.
// Wave w owns M rows [16w,16w+16) and all 64 pixels -> 4 accumulators,
// 4 WMMA per 32-wide K step. A fragments come straight from global
// (pre-swizzled); B (im2col) is staged in double-buffered LDS, n-major with
// 40-ushort rows (16B-aligned rows, conflict-free b128 fragment reads).
// Pipeline per step: barrier -> A(b128 global) -> issue next gather ->
// 4x [ds b128 + WMMA] -> commit gathered regs to other LDS buffer.
// EPI: 0 = bias, 1 = bias+ReLU, 2 = bias+ReLU+BN affine.
// ---------------------------------------------------------------------------
template <int W, bool TAP3, int EPI>
__global__ __launch_bounds__(256)
void conv_wmma_kernel(const float* __restrict__ in, float* __restrict__ out,
                      const unsigned short* __restrict__ apk,
                      const float* __restrict__ bias,
                      const float* __restrict__ bnscale,
                      const float* __restrict__ bnshift,
                      int H, int Cin, int Cout, int Kreal, int Ksteps) {
  __shared__ unsigned short ldsB[2][64 * 40];   // [buf][n][kk], rows 80B

  const int HW       = H * W;
  const int batch    = blockIdx.y;
  const int tileBase = blockIdx.x * 64;         // 64 consecutive pixels
  const float* inB = in + (size_t)batch * Cin * HW;
  float* outB      = out + (size_t)batch * Cout * HW;

  const int tid  = threadIdx.x;
  const int lane = tid & 31;
  const int wave = tid >> 5;
  const int wm   = wave << 4;

  // Per-thread im2col slots: i -> (n = (i*256+tid)&63, kk = (i*256+tid)>>6).
  // Pixel coords are K-invariant: hoist them out of the loop.
  int slot_kk[8], slot_y[8], slot_x[8], slot_off[8];
#pragma unroll
  for (int i = 0; i < 8; ++i) {
    int e = (i << 8) + tid;
    int n = e & 63;
    slot_kk[i] = e >> 6;
    int p = tileBase + n;
    slot_y[i] = p / W;                          // W = power-of-two constant
    slot_x[i] = p & (W - 1);
    slot_off[i] = n * 40 + slot_kk[i];          // LDS slot (buffer-relative)
  }

  v8f acc0 = {}, acc1 = {}, acc2 = {}, acc3 = {};

  unsigned short pend[8];
  auto gather = [&](int s) {
#pragma unroll
    for (int i = 0; i < 8; ++i) {
      int k = (s << 5) + slot_kk[i];
      unsigned short val = 0;
      if (k < Kreal) {
        int cin, dy, dx;
        if (TAP3) {
          cin = k / 9;                          // constant div -> mul-hi
          int t = k - cin * 9; int tq = t / 3;
          dy = tq - 1; dx = t - tq * 3 - 1;
        } else { cin = k; dy = 0; dx = 0; }
        int sy = slot_y[i] + dy, sx = slot_x[i] + dx;
        if (sy >= 0 && sy < H && sx >= 0 && sx < W)
          val = f2bf(inB[(size_t)cin * HW + sy * W + sx]);
      }
      pend[i] = val;
    }
  };

  // Prologue: fill buffer 0 with step 0.
  gather(0);
#pragma unroll
  for (int i = 0; i < 8; ++i) ldsB[0][slot_off[i]] = pend[i];

  const int kbase = (lane >> 4) << 3;           // 0 or 8
  const int nrow  = lane & 15;

  for (int s = 0; s < Ksteps; ++s) {
    __syncthreads();                            // buf[s&1] ready; other free
    const unsigned short* buf = ldsB[s & 1];

    // ---- A fragment: two coalesced b128 straight from global (L2) ----
    Frag16 A;
    {
      const uint4* ap = reinterpret_cast<const uint4*>(apk) +
                        (((size_t)s * 8 + wave) * 32 + lane) * 2;
      A.q[0] = ap[0];
      A.q[1] = ap[1];
    }

    // ---- issue next step's gather: loads in flight across the WMMAs ----
    if (s + 1 < Ksteps) gather(s + 1);

    // ---- B fragments (two 16B per lane: kk kbase+0..7, kbase+16..23) + 4 WMMA ----
    {
      Frag16 Bf;
      const uint4* bp = reinterpret_cast<const uint4*>(buf + (nrow + 0) * 40 + kbase);
      Bf.q[0] = bp[0]; Bf.q[1] = bp[2];
      acc0 = __builtin_amdgcn_wmma_f32_16x16x32_bf16(false, A.v, false, Bf.v,
                                                     (short)0, acc0, false, false);
    }
    {
      Frag16 Bf;
      const uint4* bp = reinterpret_cast<const uint4*>(buf + (nrow + 16) * 40 + kbase);
      Bf.q[0] = bp[0]; Bf.q[1] = bp[2];
      acc1 = __builtin_amdgcn_wmma_f32_16x16x32_bf16(false, A.v, false, Bf.v,
                                                     (short)0, acc1, false, false);
    }
    {
      Frag16 Bf;
      const uint4* bp = reinterpret_cast<const uint4*>(buf + (nrow + 32) * 40 + kbase);
      Bf.q[0] = bp[0]; Bf.q[1] = bp[2];
      acc2 = __builtin_amdgcn_wmma_f32_16x16x32_bf16(false, A.v, false, Bf.v,
                                                     (short)0, acc2, false, false);
    }
    {
      Frag16 Bf;
      const uint4* bp = reinterpret_cast<const uint4*>(buf + (nrow + 48) * 40 + kbase);
      Bf.q[0] = bp[0]; Bf.q[1] = bp[2];
      acc3 = __builtin_amdgcn_wmma_f32_16x16x32_bf16(false, A.v, false, Bf.v,
                                                     (short)0, acc3, false, false);
    }

    // ---- commit gathered regs into the other buffer (waits land here) ----
    if (s + 1 < Ksteps) {
      unsigned short* nbuf = ldsB[(s + 1) & 1];
#pragma unroll
      for (int i = 0; i < 8; ++i) nbuf[slot_off[i]] = pend[i];
    }
  }

  // ---- epilogue: bias [-> ReLU [-> BN affine]] -> store ----
#pragma unroll
  for (int r = 0; r < 8; ++r) {
    int m = wm + ((lane >> 4) << 3) + r;
    if (m < Cout) {
      float bb = bias[m];
      float v0 = acc0[r] + bb, v1 = acc1[r] + bb;
      float v2 = acc2[r] + bb, v3 = acc3[r] + bb;
      if constexpr (EPI >= 1) {
        v0 = fmaxf(v0, 0.0f); v1 = fmaxf(v1, 0.0f);
        v2 = fmaxf(v2, 0.0f); v3 = fmaxf(v3, 0.0f);
      }
      if constexpr (EPI == 2) {
        float sc = bnscale[m], sh = bnshift[m];
        v0 = v0 * sc + sh; v1 = v1 * sc + sh;
        v2 = v2 * sc + sh; v3 = v3 * sc + sh;
      }
      size_t base = (size_t)m * HW + tileBase + (lane & 15);
      outB[base]      = v0;
      outB[base + 16] = v1;
      outB[base + 32] = v2;
      outB[base + 48] = v3;
    }
  }
}

// ---------------------------------------------------------------------------
// Bilinear 2x upsample (align_corners=False), optional accumulate.
// ---------------------------------------------------------------------------
__global__ void upsample2x_kernel(const float* __restrict__ in, float* __restrict__ out,
                                  long long NC, int Hin, int Win, int accumulate) {
  int Hout = Hin * 2, Wout = Win * 2;
  long long idx = (long long)blockIdx.x * blockDim.x + threadIdx.x;
  long long total = NC * Hout * Wout;
  if (idx >= total) return;
  int x = (int)(idx % Wout);
  long long t = idx / Wout;
  int y = (int)(t % Hout);
  long long c = t / Hout;
  float fy = (y + 0.5f) * 0.5f - 0.5f;
  float fx = (x + 0.5f) * 0.5f - 0.5f;
  int y0 = (int)floorf(fy), x0 = (int)floorf(fx);
  float wy = fy - (float)y0, wx = fx - (float)x0;
  int y0c = min(max(y0, 0), Hin - 1), y1c = min(max(y0 + 1, 0), Hin - 1);
  int x0c = min(max(x0, 0), Win - 1), x1c = min(max(x0 + 1, 0), Win - 1);
  const float* p = in + (size_t)c * Hin * Win;
  float v = (1.0f - wy) * ((1.0f - wx) * p[y0c * Win + x0c] + wx * p[y0c * Win + x1c])
          +        wy  * ((1.0f - wx) * p[y1c * Win + x0c] + wx * p[y1c * Win + x1c]);
  if (accumulate) out[idx] += v; else out[idx] = v;
}

// ---------------------------------------------------------------------------
// Copy agg [B][128][HW] into concat [B][178][HW] channels 0..127
// ---------------------------------------------------------------------------
__global__ void copy_agg_kernel(const float* __restrict__ agg, float* __restrict__ concat,
                                int HW) {
  long long idx = (long long)blockIdx.x * blockDim.x + threadIdx.x;
  long long per = (long long)128 * HW;
  long long total = 8LL * per;
  if (idx >= total) return;
  long long b = idx / per;
  long long r = idx - b * per;
  concat[(size_t)b * 178 * HW + r] = agg[idx];
}

// ---------------------------------------------------------------------------
// Gaussian soft-attention masks -> concat channels 128..177.
// One workgroup per (batch, box); LDS reduce for normalization.
// ---------------------------------------------------------------------------
__global__ __launch_bounds__(256)
void attention_kernel(const float* __restrict__ boxes, float* __restrict__ concat,
                      int H, int W) {
  __shared__ float red[256];
  int bn = blockIdx.x;            // 0 .. B*50-1
  int b = bn / 50, n = bn - b * 50;
  const float* bx = boxes + (size_t)bn * 4;
  float mu0 = floorf(bx[0] * 0.125f);
  float mu1 = floorf(bx[1] * 0.125f);
  float bd0 = floorf(bx[2] * 0.125f);
  float bd1 = floorf(bx[3] * 0.125f);
  float v0 = bd0 + 1e-7f, v1 = bd1 + 1e-7f;
  float norm = 1.0f / (6.28318530717958647692f * sqrtf(v0 * v1));
  float hy = floorf(bd1 * 0.5f);       // quirky axis swap per reference
  float hx = floorf(bd0 * 0.5f);
  float y1 = mu0 - hy, y2 = mu0 + hy;
  float x1 = mu1 - hx, x2 = mu1 + hx;
  int HW = H * W;
  float* dst = concat + ((size_t)b * 178 + 128 + n) * HW;

  float sum = 0.0f;
  for (int p = threadIdx.x; p < HW; p += 256) {
    int y = p / W, x = p - (p / W) * W;
    float fy = (float)y, fx = (float)x;
    if (fy >= y1 && fy < y2 && fx >= x1 && fx < x2) {
      float dy = fy - mu0, dx = fx - mu1;
      sum += norm * __expf(-0.5f * (dy * dy / v0 + dx * dx / v1));
    }
  }
  red[threadIdx.x] = sum;
  __syncthreads();
  for (int s = 128; s > 0; s >>= 1) {
    if ((int)threadIdx.x < s) red[threadIdx.x] += red[threadIdx.x + s];
    __syncthreads();
  }
  float inv = 1.0f / (red[0] + 1e-7f);
  for (int p = threadIdx.x; p < HW; p += 256) {
    int y = p / W, x = p - (p / W) * W;
    float fy = (float)y, fx = (float)x;
    float g = 0.0f;
    if (fy >= y1 && fy < y2 && fx >= x1 && fx < x2) {
      float dy = fy - mu0, dx = fx - mu1;
      g = norm * __expf(-0.5f * (dy * dy / v0 + dx * dx / v1));
    }
    dst[p] = g * inv;
  }
}

// ---------------------------------------------------------------------------
// Host launcher
// ---------------------------------------------------------------------------
extern "C" void kernel_launch(void* const* d_in, const int* in_sizes, int n_in,
                              void* d_out, int out_size, void* d_ws, size_t ws_size,
                              hipStream_t stream) {
  const int B = 8;
  (void)out_size; (void)ws_size;

  // ------ resolve flattened input ordering ------
  const float *f3, *f4, *f5, *boxes;
  const float *w_up3, *b_up3, *w_up4, *b_up4, *w_up5a, *b_up5a, *w_up5b, *b_up5b;
  const float *w_pre, *b_pre, *pre_g, *pre_b, *pre_m, *pre_v;
  const float *w_pan[4], *b_pan[4], *pan_g[4], *pan_b[4], *pan_m[4], *pan_v[4];
  const float *w_out, *b_out;
  const float* const* in = (const float* const*)d_in;

  if (n_in >= 44 && in_sizes[0] == 1600) {
    // jax tree_flatten (alphabetical dict keys)
    boxes = in[0]; f3 = in[1]; f4 = in[2]; f5 = in[3];
    b_out = in[4]; w_out = in[5];
    int i = 6;
    for (int l = 0; l < 4; ++l) {
      b_pan[l] = in[i++]; pan_b[l] = in[i++]; pan_g[l] = in[i++];
      pan_m[l] = in[i++]; pan_v[l] = in[i++]; w_pan[l] = in[i++];
    }
    b_pre = in[i++]; pre_b = in[i++]; pre_g = in[i++];
    pre_m = in[i++]; pre_v = in[i++]; w_pre = in[i++];
    b_up3 = in[i++]; w_up3 = in[i++];
    b_up4 = in[i++]; w_up4 = in[i++];
    b_up5a = in[i++]; w_up5a = in[i++];
    b_up5b = in[i++]; w_up5b = in[i++];
  } else {
    // insertion order fallback
    int i = 0;
    f3 = in[i++]; f4 = in[i++]; f5 = in[i++]; boxes = in[i++];
    w_up3 = in[i++]; b_up3 = in[i++];
    w_up4 = in[i++]; b_up4 = in[i++];
    w_up5a = in[i++]; b_up5a = in[i++];
    w_up5b = in[i++]; b_up5b = in[i++];
    w_pre = in[i++]; b_pre = in[i++];
    pre_g = in[i++]; pre_b = in[i++]; pre_m = in[i++]; pre_v = in[i++];
    for (int l = 0; l < 4; ++l) {
      w_pan[l] = in[i++]; b_pan[l] = in[i++];
      pan_g[l] = in[i++]; pan_b[l] = in[i++]; pan_m[l] = in[i++]; pan_v[l] = in[i++];
    }
    w_out = in[i++]; b_out = in[i++];
  }

  // ------ workspace bump allocator ------
  char* ws = (char*)d_ws;
  size_t off = 0;
  auto alloc = [&](size_t bytes) -> char* {
    char* p = ws + off;
    off = (off + bytes + 255) & ~(size_t)255;
    return p;
  };

  const int KS_256 = 72;   // 256*9 / 32
  const int KS_128 = 36;   // 128*9 / 32
  const int KS_pre = 51;   // ceil(178*9 / 32)
  const int KS_out = 4;    // 128 / 32
  const int K_256 = 2304, K_128 = 1152, K_pre = 1602, K_out = 128;

  unsigned short* Wup3  = (unsigned short*)alloc((size_t)KS_256 * 4096 * 2);
  unsigned short* Wup4  = (unsigned short*)alloc((size_t)KS_256 * 4096 * 2);
  unsigned short* Wup5a = (unsigned short*)alloc((size_t)KS_256 * 4096 * 2);
  unsigned short* Wup5b = (unsigned short*)alloc((size_t)KS_128 * 4096 * 2);
  unsigned short* Wpre  = (unsigned short*)alloc((size_t)KS_pre * 4096 * 2);
  unsigned short* Wpan[4];
  for (int l = 0; l < 4; ++l) Wpan[l] = (unsigned short*)alloc((size_t)KS_128 * 4096 * 2);
  unsigned short* Wout  = (unsigned short*)alloc((size_t)KS_out * 4096 * 2);

  float* sc_pre = (float*)alloc(128 * 4);
  float* sh_pre = (float*)alloc(128 * 4);
  float* sc_pan[4]; float* sh_pan[4];
  for (int l = 0; l < 4; ++l) { sc_pan[l] = (float*)alloc(128 * 4); sh_pan[l] = (float*)alloc(128 * 4); }

  float* t5a    = (float*)alloc((size_t)B * 128 * 32 * 32 * 4);
  float* t5au   = (float*)alloc((size_t)B * 128 * 64 * 64 * 4);
  float* t5b    = (float*)alloc((size_t)B * 128 * 64 * 64 * 4);
  float* t4     = (float*)alloc((size_t)B * 128 * 64 * 64 * 4);
  float* t3     = (float*)alloc((size_t)B * 128 * 128 * 128 * 4);
  float* concat = (float*)alloc((size_t)B * 178 * 128 * 128 * 4);
  float* x1     = (float*)alloc((size_t)B * 128 * 128 * 128 * 4);
  float* x2     = (float*)alloc((size_t)B * 128 * 128 * 128 * 4);

  // ------ pack weights (A-fragment layout) + BN precompute ------
  auto pack = [&](const float* w, unsigned short* dst, int Kreal, int Cout, int Ksteps) {
    int total = Ksteps * 4096;
    pack_weights_kernel<<<(total + 255) / 256, 256, 0, stream>>>(w, dst, Kreal, Cout, total);
  };
  pack(w_up3,  Wup3,  K_256, 128, KS_256);
  pack(w_up4,  Wup4,  K_256, 128, KS_256);
  pack(w_up5a, Wup5a, K_256, 128, KS_256);
  pack(w_up5b, Wup5b, K_128, 128, KS_128);
  pack(w_pre,  Wpre,  K_pre, 128, KS_pre);
  for (int l = 0; l < 4; ++l) pack(w_pan[l], Wpan[l], K_128, 128, KS_128);
  pack(w_out,  Wout,  K_out, 52, KS_out);

  bn_prep_kernel<<<1, 128, 0, stream>>>(pre_g, pre_b, pre_m, pre_v, sc_pre, sh_pre, 128);
  for (int l = 0; l < 4; ++l)
    bn_prep_kernel<<<1, 128, 0, stream>>>(pan_g[l], pan_b[l], pan_m[l], pan_v[l],
                                          sc_pan[l], sh_pan[l], 128);

  // ------ FPN aggregation (bias+ReLU epilogue) ------
  conv_wmma_kernel<32, true, 1><<<dim3(32 * 32 / 64, B), 256, 0, stream>>>(
      f5, t5a, Wup5a, b_up5a, nullptr, nullptr, 32, 256, 128, K_256, KS_256);
  {
    long long total = (long long)B * 128 * 64 * 64;
    upsample2x_kernel<<<(int)((total + 255) / 256), 256, 0, stream>>>(
        t5a, t5au, (long long)B * 128, 32, 32, 0);
  }
  conv_wmma_kernel<64, true, 1><<<dim3(64 * 64 / 64, B), 256, 0, stream>>>(
      t5au, t5b, Wup5b, b_up5b, nullptr, nullptr, 64, 128, 128, K_128, KS_128);
  conv_wmma_kernel<64, true, 1><<<dim3(64 * 64 / 64, B), 256, 0, stream>>>(
      f4, t4, Wup4, b_up4, nullptr, nullptr, 64, 256, 128, K_256, KS_256);
  conv_wmma_kernel<128, true, 1><<<dim3(128 * 128 / 64, B), 256, 0, stream>>>(
      f3, t3, Wup3, b_up3, nullptr, nullptr, 128, 256, 128, K_256, KS_256);
  {
    long long total = (long long)B * 128 * 128 * 128;
    upsample2x_kernel<<<(int)((total + 255) / 256), 256, 0, stream>>>(
        t5b, t3, (long long)B * 128, 64, 64, 1);
    upsample2x_kernel<<<(int)((total + 255) / 256), 256, 0, stream>>>(
        t4, t3, (long long)B * 128, 64, 64, 1);
  }

  // ------ concat(agg, attention) ------
  {
    long long total = (long long)B * 128 * 128 * 128;
    copy_agg_kernel<<<(int)((total + 255) / 256), 256, 0, stream>>>(t3, concat, 128 * 128);
  }
  attention_kernel<<<B * 50, 256, 0, stream>>>(boxes, concat, 128, 128);

  // ------ panoptic head (bias+ReLU+BN epilogue) ------
  conv_wmma_kernel<128, true, 2><<<dim3(128 * 128 / 64, B), 256, 0, stream>>>(
      concat, x1, Wpre, b_pre, sc_pre, sh_pre, 128, 178, 128, K_pre, KS_pre);
  conv_wmma_kernel<128, true, 2><<<dim3(128 * 128 / 64, B), 256, 0, stream>>>(
      x1, x2, Wpan[0], b_pan[0], sc_pan[0], sh_pan[0], 128, 128, 128, K_128, KS_128);
  conv_wmma_kernel<128, true, 2><<<dim3(128 * 128 / 64, B), 256, 0, stream>>>(
      x2, x1, Wpan[1], b_pan[1], sc_pan[1], sh_pan[1], 128, 128, 128, K_128, KS_128);
  conv_wmma_kernel<128, true, 2><<<dim3(128 * 128 / 64, B), 256, 0, stream>>>(
      x1, x2, Wpan[2], b_pan[2], sc_pan[2], sh_pan[2], 128, 128, 128, K_128, KS_128);
  conv_wmma_kernel<128, true, 2><<<dim3(128 * 128 / 64, B), 256, 0, stream>>>(
      x2, x1, Wpan[3], b_pan[3], sc_pan[3], sh_pan[3], 128, 128, 128, K_128, KS_128);

  // ------ output 1x1 conv (bias only) ------
  conv_wmma_kernel<128, false, 0><<<dim3(128 * 128 / 64, B), 256, 0, stream>>>(
      x1, (float*)d_out, Wout, b_out, nullptr, nullptr, 128, 128, 52, K_out, KS_out);
}
